// GINetConvLayer_65403761983528
// MI455X (gfx1250) — matrix operand order repaired and checked by reading, not compiled, not run on hardware
//
#include <hip/hip_runtime.h>

// ---------------------------------------------------------------------------
// GINetConv-like layer for MI455X (gfx1250, wave32, WMMA).
// Dominant cost: msg[E,1806] @ W^T[1806,64]  (~22.2 GFLOP).
// Strategy: never materialize msg in HBM. Build 16-edge msg tiles in LDS
// (fp16, K padded to 1824 = 57*32) and contract with v_wmma_f32_16x16x32_f16.
// edge_f (E*129 fp32 = 50MB) stays L2-resident (192MB L2) for the gathers.
// ---------------------------------------------------------------------------

typedef __attribute__((ext_vector_type(16))) _Float16 v16h;
typedef __attribute__((ext_vector_type(8)))  float    v8f;

#define NN    6000
#define KK    16
#define CIN   64
#define COUT  64
#define AA    7
#define FF    129              // 2*CIN + 1
#define EE    (NN*KK)          // 96000
#define FDIM  (AA*2*FF)        // 1806
#define NKC   57               // ceil(1806/32) -> K padded to 1824
#define EF_STRIDE 132          // edge_f row stride (floats, 16B aligned)
#define A_STRIDE_H 1832        // msg LDS row stride in halfs (916 dwords; 916%64=20 -> conflict-free rows)
#define A_STRIDE_W 916         // in dwords

// ---------------- prep kernels ----------------

__global__ void k_zero_out(float* out) {
  int i = blockIdx.x * blockDim.x + threadIdx.x;
  if (i < NN * COUT) out[i] = 0.0f;
}

// edge_f[e] = [ x[row[e]] (64) | x[col[e]] (64) | edge_attr[e] (1) | pad(3) ]
__global__ void k_edge_f(const float* __restrict__ x, const int* __restrict__ ei,
                         const float* __restrict__ eattr, float* __restrict__ ef) {
  int e = blockIdx.x * blockDim.x + threadIdx.x;
  if (e >= EE) return;
  int r = ei[e], c = ei[EE + e];
  const float4* xr = (const float4*)(x + (size_t)r * CIN);
  const float4* xc = (const float4*)(x + (size_t)c * CIN);
  float4* o = (float4*)(ef + (size_t)e * EF_STRIDE);
#pragma unroll
  for (int i = 0; i < 16; ++i) o[i] = xr[i];
#pragma unroll
  for (int i = 0; i < 16; ++i) o[16 + i] = xc[i];
  float* of = ef + (size_t)e * EF_STRIDE;
  of[128] = eattr[e];
  of[129] = 0.0f; of[130] = 0.0f; of[131] = 0.0f;
}

// col is K stacked permutations of [0,N): node n has exactly one in-edge with
// e%K == j for each j -> deterministic slot assignment, no atomics needed.
// (k-order within a node only permutes an order-invariant sum/max.)
__global__ void k_inedges(const int* __restrict__ ei, int* __restrict__ in_edges) {
  int e = blockIdx.x * blockDim.x + threadIdx.x;
  if (e >= EE) return;
  int c = ei[EE + e];
  in_edges[c * KK + (e & (KK - 1))] = e;
}

// neighbor edge ids + angle bins for every (edge, k)
__global__ void k_bins(const int* __restrict__ ei, const float* __restrict__ pos,
                       const int* __restrict__ in_edges,
                       int* __restrict__ nbrE, unsigned char* __restrict__ bins) {
  int t = blockIdx.x * blockDim.x + threadIdx.x;
  if (t >= EE * KK) return;
  int e = t >> 4, k = t & 15;
  int r = ei[e], c = ei[EE + e];
  int ne = in_edges[r * KK + k];
  nbrE[t] = ne;
  int nr = ei[ne];                               // src of incoming edge
  float rx = pos[r*3], ry = pos[r*3+1], rz = pos[r*3+2];
  float v1x = rx - pos[nr*3], v1y = ry - pos[nr*3+1], v1z = rz - pos[nr*3+2];
  float v2x = pos[c*3] - rx,  v2y = pos[c*3+1] - ry, v2z = pos[c*3+2] - rz;
  float dot = v1x*v2x + v1y*v2y + v1z*v2z;
  float n1 = sqrtf(v1x*v1x + v1y*v1y + v1z*v1z);
  float n2 = sqrtf(v2x*v2x + v2y*v2y + v2z*v2z);
  float ca = dot / (n1 * n2 + 1e-8f);
  ca = fminf(1.0f, fmaxf(-1.0f, ca));
  float ang = acosf(ca);
  int b = (int)(ang * ((float)AA / 3.14159265358979323846f));
  b = b < 0 ? 0 : (b > AA - 1 ? AA - 1 : b);
  bins[t] = (unsigned char)b;
}

// Pack W (fp32 [64,1806]) into fp16 WMMA B fragments, K padded to 1824.
// B fragment layout used here (32x16, 16-bit, contiguous k order matching the A reads):
//   lane n = L&15 (column), K-half = L>>4 (K 0..15 vs 16..31), VGPR j -> K = 16*(L>>4)+2j,+1.
// Stored so each lane's 8 dwords are contiguous: W16[((kc*4+nt)*32 + L)*8 + j]
__global__ void k_packW(const float* __restrict__ W, unsigned int* __restrict__ W16) {
  int t = blockIdx.x * blockDim.x + threadIdx.x;
  if (t >= NKC * 4 * 32 * 8) return;
  int j    =  t        & 7;
  int lane = (t >> 3)  & 31;
  int nt   = (t >> 8)  & 3;
  int kc   =  t >> 10;
  int n  = nt * 16 + (lane & 15);
  int kk = ((lane >> 4) << 4) + j * 2;
  int k0 = kc * 32 + kk, k1 = k0 + 1;
  float w0 = (k0 < FDIM) ? W[(size_t)n * FDIM + k0] : 0.0f;
  float w1 = (k1 < FDIM) ? W[(size_t)n * FDIM + k1] : 0.0f;
  union { _Float16 h[2]; unsigned int u; } pk;
  pk.h[0] = (_Float16)w0;
  pk.h[1] = (_Float16)w1;
  W16[t] = pk.u;
}

// ---------------- fused aggregate + GEMM + scatter ----------------
// One WG per 16-edge tile. 4 waves.
// Phase 1: all 128 threads build the 16x1806 msg tile in LDS (fp32 acc -> fp16).
// Phase 2: wave w computes output channels [16w,16w+16): 57 WMMAs over K=1824.
// Epilogue: atomic scatter-add into out[col[e]].
__global__ __launch_bounds__(128) void k_main(
    const float* __restrict__ ef, const int* __restrict__ nbrE,
    const unsigned char* __restrict__ bins, const int* __restrict__ ei,
    const unsigned int* __restrict__ W16, const float* __restrict__ bmsg,
    float* __restrict__ out) {
  __shared__ unsigned int  smsg[16 * A_STRIDE_W];   // 16 rows x 1832 halfs (58.6 KB)
  __shared__ int           s_nbr[16 * KK];
  __shared__ unsigned char s_bin[16 * KK];
  _Float16* sh = (_Float16*)smsg;

  const int tid   = threadIdx.x;
  const int tileE = blockIdx.x * 16;

  // stage neighbor ids & bins for this tile; zero the K-padding columns
  for (int i = tid; i < 16 * KK; i += 128) {
    s_nbr[i] = nbrE[(size_t)tileE * KK + i];
    s_bin[i] = bins[(size_t)tileE * KK + i];
  }
  for (int i = tid; i < 16 * (A_STRIDE_H - FDIM); i += 128) {
    int e = i / (A_STRIDE_H - FDIM), c = FDIM + i % (A_STRIDE_H - FDIM);
    sh[e * A_STRIDE_H + c] = (_Float16)0.0f;
  }
  __syncthreads();

  // ---- phase 1: per-(edge,feature) bin-wise sum & max over K=16 neighbors ----
  for (int it = tid; it < 16 * FF; it += 128) {
    int el = it / FF, f = it - el * FF;
    float s[AA], m[AA];
#pragma unroll
    for (int a = 0; a < AA; ++a) { s[a] = 0.0f; m[a] = -3.4028235e38f; }
#pragma unroll
    for (int k = 0; k < KK; ++k) {
      int   ne = s_nbr[el * KK + k];
      int   b  = s_bin[el * KK + k];
      float v  = ef[(size_t)ne * EF_STRIDE + f];       // L2-resident gather
#pragma unroll
      for (int a = 0; a < AA; ++a) {
        bool hit = (b == a);
        s[a] += hit ? v : 0.0f;
        m[a]  = hit ? fmaxf(m[a], v) : m[a];
      }
    }
#pragma unroll
    for (int a = 0; a < AA; ++a) {
      float mx = (m[a] < -3.0e38f) ? 0.0f : m[a];      // empty bin -> 0 (ref semantics)
      sh[el * A_STRIDE_H + a * (2 * FF) + f]      = (_Float16)s[a];
      sh[el * A_STRIDE_H + a * (2 * FF) + FF + f] = (_Float16)mx;
    }
  }
  __syncthreads();

  // ---- phase 2: GEMM, wave nt owns channels [16*nt, 16*nt+16) ----
  const int lane = tid & 31;
  const int nt   = tid >> 5;
  const int m16  = lane & 15;
  const int hh   = lane >> 4;
  const int arow = m16 * A_STRIDE_W;                   // dwords
  const uint4* wq = (const uint4*)W16;

  v8f acc = {};
#pragma unroll 4
  for (int kc = 0; kc < NKC; ++kc) {
    // A fragment (16x32 f16): lane m=L&15, half h=L>>4;
    // dword offsets within the 16-dword K-chunk: h=0 -> {0..7}, h=1 -> {8..15}
    union { v16h v; unsigned int u[8]; } afr;
    const int kbase = kc * 16;                         // dwords
#pragma unroll
    for (int j = 0; j < 8; ++j) {
      int bd = j + 8 * hh;                             // {0..7} (+8 if hh)
      afr.u[j] = smsg[arow + kbase + bd];
    }
    // B fragment: contiguous 32B per lane from pre-packed W
    union { v16h v; uint4 q[2]; } bfr;
    size_t bidx = ((size_t)(kc * 4 + nt) * 32 + lane) * 2;
    bfr.q[0] = wq[bidx];
    bfr.q[1] = wq[bidx + 1];

    acc = __builtin_amdgcn_wmma_f32_16x16x32_f16(
        false, afr.v, false, bfr.v, (short)0, acc, false, false);
  }

  // ---- epilogue: D layout lane n=L&15, VGPR j -> M = j + 8*(L>>4) ----
  const int   ch   = nt * 16 + m16;
  const float bias = bmsg[ch];
#pragma unroll
  for (int j = 0; j < 8; ++j) {
    int em   = j + 8 * hh;
    int node = ei[EE + tileE + em];                    // col[e]
    atomicAdd(&out[(size_t)node * COUT + ch], acc[j] + bias);
  }
}

// ---------------- host launcher ----------------

extern "C" void kernel_launch(void* const* d_in, const int* in_sizes, int n_in,
                              void* d_out, int out_size, void* d_ws, size_t ws_size,
                              hipStream_t stream) {
  const float* x     = (const float*)d_in[0];   // [N,64]
  const int*   ei    = (const int*)  d_in[1];   // [2,E]
  const float* eattr = (const float*)d_in[2];   // [E,1]
  const float* pos   = (const float*)d_in[3];   // [N,3]
  const float* W     = (const float*)d_in[4];   // [64,1806]
  const float* bmsg  = (const float*)d_in[5];   // [64]
  float*       out   = (float*)d_out;           // [N,64]

  char* ws = (char*)d_ws;
  size_t off = 0;
  float* ef        = (float*)(ws + off);        off += (size_t)EE * EF_STRIDE * 4;  // 50.7 MB
  int*   in_edges  = (int*)(ws + off);          off += (size_t)NN * KK * 4;
  int*   nbrE      = (int*)(ws + off);          off += (size_t)EE * KK * 4;
  unsigned int* W16 = (unsigned int*)(ws + off); off += (size_t)NKC * 4 * 32 * 8 * 4;
  unsigned char* bins = (unsigned char*)(ws + off); off += (size_t)EE * KK;
  (void)ws_size; (void)in_sizes; (void)n_in; (void)out_size;

  k_zero_out<<<(NN * COUT + 255) / 256, 256, 0, stream>>>(out);
  k_packW   <<<(NKC * 4 * 32 * 8 + 255) / 256, 256, 0, stream>>>(W, W16);
  k_edge_f  <<<(EE + 255) / 256, 256, 0, stream>>>(x, ei, eattr, ef);
  k_inedges <<<(EE + 255) / 256, 256, 0, stream>>>(ei, in_edges);
  k_bins    <<<(EE * KK + 255) / 256, 256, 0, stream>>>(ei, pos, in_edges, nbrE, bins);
  k_main    <<<EE / 16, 128, 0, stream>>>(ef, nbrE, bins, ei, W16, bmsg, out);
}
